// MyModel_73469710565761
// MI455X (gfx1250) — matrix operand order repaired and checked by reading, not compiled, not run on hardware
//
#include <hip/hip_runtime.h>

// ---------------------------------------------------------------------------
// BigBird-style sparse-attention BERT forward for gfx1250 (MI455X, wave32).
// Matmuls on v_wmma_f32_16x16x32_bf16; LDS staging via async global->LDS DMA.
// ---------------------------------------------------------------------------

typedef __bf16 bf16;
typedef __bf16  v16bf __attribute__((ext_vector_type(16)));
typedef float   v8f   __attribute__((ext_vector_type(8)));
typedef unsigned short u16x8 __attribute__((ext_vector_type(8)));

#define WMMA_BF16(A, B, C) \
    __builtin_amdgcn_wmma_f32_16x16x32_bf16(false, (A), false, (B), (short)0, (C), false, false)

// Async 16-byte global->LDS copy (GLOBAL_LOAD_ASYNC_TO_LDS_B128, ASYNCcnt).
// For a flat address in the LDS aperture, bits[31:0] are the LDS byte offset,
// so the low 32 bits of a generic pointer to __shared__ are the VDST operand.
__device__ inline void async_copy16(void* lds_dst, const void* gsrc) {
    unsigned d = (unsigned)(size_t)lds_dst;
    unsigned long long a = (unsigned long long)(size_t)gsrc;
    asm volatile("global_load_async_to_lds_b128 %0, %1, off"
                 :: "v"(d), "v"(a) : "memory");
}
__device__ inline void async_wait0() {
    asm volatile("s_wait_asynccnt 0x0" ::: "memory");
}

// ===========================================================================
// LDS fragment layouts (bf16, 16x16x32 WMMA):
// A-fragment (16x32, MxK): lane = (M&15) + 16*(K>>4)     ; e = K & 15
// B-fragment (32x16, KxN): lane = (N&15) + 16*(K>>4)     ; e = K & 15
// Fragment = 32 lanes x 16 bf16 = 1 KB; register load = 32B read at frag[lane*16].
// ===========================================================================

// Stage 128x32 row-major bf16 tile as 8 A-fragments (async).
__device__ inline void stage_A(const bf16* __restrict__ src, int ld,
                               bf16 (*frag)[512], int tid /*0..255*/) {
    int r  = tid >> 1;
    int kh = tid & 1;
    const bf16* g = src + (size_t)r * ld + kh * 16;
    int f  = r >> 4;
    int lr = r & 15;
    async_copy16(&frag[f][(lr + 16 * kh) * 16], g);         // lane lr+16*kh (K lo 8)
    async_copy16(&frag[f][(lr + 16 * kh) * 16 + 8], g + 8); // lane lr+16*kh (K hi 8)
}

// Stage 128x32 tile of Wt (row-major [N][K]) as 8 B-fragments (async).
__device__ inline void stage_B(const bf16* __restrict__ src, int ld,
                               bf16 (*frag)[512], int tid /*0..255*/) {
    int n  = tid >> 1;
    int kh = tid & 1;
    const bf16* g = src + (size_t)n * ld + kh * 16;
    int f    = n >> 4;
    int lane = (n & 15) + 16 * kh;
    async_copy16(&frag[f][lane * 16], g);
    async_copy16(&frag[f][lane * 16 + 8], g + 8);
}

// ===========================================================================
// GEMM: C[M,N] = A[M,K](bf16 rm) * Wt[N,K](bf16) + bias, optional tanh-GELU.
// 256 threads (8 waves), tile 128x128, K-step 32.
// ===========================================================================
__global__ __launch_bounds__(256) void gemm_bf16_wmma(
    const bf16* __restrict__ A, const bf16* __restrict__ Wt,
    const float* __restrict__ bias, bf16* __restrict__ outB,
    float* __restrict__ outF, int M, int N, int K, int dogelu)
{
    __shared__ bf16 sA[8][512];
    __shared__ bf16 sB[8][512];
    int tid  = threadIdx.x;
    int w    = tid >> 5;
    int lane = tid & 31;
    int m0 = blockIdx.y * 128;
    int n0 = blockIdx.x * 128;

    v8f acc[8] = {};

    for (int k0 = 0; k0 < K; k0 += 32) {
        stage_A(A  + (size_t)m0 * K + k0, K, sA, tid);
        stage_B(Wt + (size_t)n0 * K + k0, K, sB, tid);
        if (k0 + 32 < K) {  // software prefetch of next K tiles -> global_prefetch_b8
            __builtin_prefetch(A  + (size_t)(m0 + (tid >> 1)) * K + k0 + 32 + (tid & 1) * 16, 0, 1);
            __builtin_prefetch(Wt + (size_t)(n0 + (tid >> 1)) * K + k0 + 32 + (tid & 1) * 16, 0, 1);
        }
        async_wait0();
        __syncthreads();
        v16bf a = *(const v16bf*)&sA[w][lane * 16];
        v16bf bfrag[8];
        #pragma unroll
        for (int nt = 0; nt < 8; nt++)
            bfrag[nt] = *(const v16bf*)&sB[nt][lane * 16];
        #pragma unroll
        for (int nt = 0; nt < 8; nt++)
            acc[nt] = WMMA_BF16(a, bfrag[nt], acc[nt]);
        __syncthreads();
    }

    // Epilogue. C layout: lanes 0-15 -> N=lane, M=e ; lanes 16-31 -> M=e+8.
    int lcol  = lane & 15;
    int rbase = m0 + w * 16 + 8 * (lane >> 4);
    #pragma unroll
    for (int nt = 0; nt < 8; nt++) {
        int c    = n0 + nt * 16 + lcol;
        float bb = bias ? bias[c] : 0.0f;
        #pragma unroll
        for (int e = 0; e < 8; e++) {
            float v = acc[nt][e] + bb;
            if (dogelu) {
                float x = v;
                v = 0.5f * x * (1.0f + tanhf(0.79788456f * (x + 0.044715f * x * x * x)));
            }
            size_t off = (size_t)(rbase + e) * N + c;
            if (outB) outB[off] = (bf16)v;
            if (outF) outF[off] = v;
        }
    }
}

// ===========================================================================
// Block-sparse attention, one workgroup (128 thr / 4 waves) per (b,h,qblock).
// Flash-style online softmax over the 8 selected 64-key blocks.
// ===========================================================================
__global__ __launch_bounds__(128) void bigbird_attn(
    const bf16* __restrict__ qkv,      // [16384][2304] bf16 (q|k|v)
    const int*  __restrict__ amask,    // [B*S]
    const int*  __restrict__ rand_idx, // [64][3]
    bf16* __restrict__ obuf)           // [16384][768]
{
    __shared__ bf16 sQ[8][512];        // 8 A-frags: (rowtile 0..3) x (dtile 0..1)
    __shared__ bf16 sK[8][512];        // 8 B-frags for Q*K^T: (coltile 0..3) x (dtile 0..1)
    __shared__ bf16 sV[8][512];        // 8 B-frags for P*V:   (ntile 0..3) x (ktile 0..1)
    __shared__ bf16 sP[4][2][512];     // per-wave P as 2 A-frags (K=64 split in 2)

    int tid = threadIdx.x, w = tid >> 5, lane = tid & 31;
    int n  = blockIdx.x % 64;
    int hh = (blockIdx.x / 64) % 12;
    int b  = blockIdx.x / (64 * 12);
    size_t tok0 = (size_t)(b * 64 + n) * 64;

    // ---- stage Q once (A-fragments, async) ----
    for (int c = tid; c < 256; c += 128) {
        int r = c >> 2, dc = c & 3;
        int dt = dc >> 1, kh = dc & 1;
        const bf16* g = qkv + (tok0 + r) * 2304 + hh * 64 + dc * 16;
        int f = (r >> 4) * 2 + dt;
        async_copy16(&sQ[f][((r & 15) + 16 * kh) * 16], g);
        async_copy16(&sQ[f][((r & 15) + 16 * kh) * 16 + 8], g + 8);
    }
    async_wait0();
    __syncthreads();
    v16bf qa0 = *(const v16bf*)&sQ[w * 2 + 0][lane * 16];
    v16bf qa1 = *(const v16bf*)&sQ[w * 2 + 1][lane * 16];

    float m_run[8], l_run[8];
    #pragma unroll
    for (int e = 0; e < 8; e++) { m_run[e] = -1e30f; l_run[e] = 0.0f; }
    v8f oacc[4] = {};

    const float scale = 0.125f;   // 1/sqrt(64)

    for (int j = 0; j < 8; j++) {
        int selb;
        if      (j == 0) selb = 0;
        else if (j == 1) selb = (n + 63) & 63;
        else if (j == 2) selb = n;
        else if (j == 3) selb = (n + 1) & 63;
        else if (j == 4) selb = 63;
        else             selb = rand_idx[n * 3 + (j - 5)] & 63;
        size_t ktok = (size_t)(b * 64 + selb) * 64;

        // ---- stage K block as B-fragments (columns = keys, K-dim = d), async ----
        for (int c = tid; c < 256; c += 128) {
            int key = c >> 2, dc = c & 3;
            int dt = dc >> 1, kh = dc & 1;
            const bf16* g = qkv + (ktok + key) * 2304 + 768 + hh * 64 + dc * 16;
            int f   = (key >> 4) * 2 + dt;
            int lnn = (key & 15) + 16 * kh;
            async_copy16(&sK[f][lnn * 16], g);
            async_copy16(&sK[f][lnn * 16 + 8], g + 8);
        }
        // ---- stage V block as B-fragments (columns = d, K-dim = keys) ----
        for (int c = tid; c < 256; c += 128) {
            int key = c >> 2, nt = c & 3;
            const bf16* g = qkv + (ktok + key) * 2304 + 1536 + hh * 64 + nt * 16;
            int kt = key >> 5, kk = key & 31;
            int e  = kk & 15;
            int lb = 16 * (kk >> 4);
            bf16* dst = &sV[nt * 2 + kt][0];
            #pragma unroll
            for (int jj = 0; jj < 16; jj++) dst[(lb + jj) * 16 + e] = g[jj];
        }
        async_wait0();
        __syncthreads();

        // ---- scores: 16 q rows x 64 keys per wave ----
        v8f sc[4];
        #pragma unroll
        for (int ct = 0; ct < 4; ct++) {
            v8f z = {};
            z = WMMA_BF16(qa0, *(const v16bf*)&sK[ct * 2 + 0][lane * 16], z);
            z = WMMA_BF16(qa1, *(const v16bf*)&sK[ct * 2 + 1][lane * 16], z);
            sc[ct] = z;
        }

        float mx[8];
        #pragma unroll
        for (int e = 0; e < 8; e++) mx[e] = -3e38f;
        #pragma unroll
        for (int ct = 0; ct < 4; ct++) {
            int key = ct * 16 + (lane & 15);
            float biasv = (amask[ktok + key] > 0) ? 0.0f : -1e9f;
            #pragma unroll
            for (int e = 0; e < 8; e++) {
                float v = sc[ct][e] * scale + biasv;
                sc[ct][e] = v;
                mx[e] = fmaxf(mx[e], v);
            }
        }
        // row-max over the 16-lane group holding this row
        #pragma unroll
        for (int mk = 1; mk < 16; mk <<= 1)
            #pragma unroll
            for (int e = 0; e < 8; e++) mx[e] = fmaxf(mx[e], __shfl_xor(mx[e], mk, 32));

        float alpha[8], mnew[8], rs[8];
        #pragma unroll
        for (int e = 0; e < 8; e++) {
            mnew[e]  = fmaxf(m_run[e], mx[e]);
            alpha[e] = __expf(m_run[e] - mnew[e]);
            rs[e]    = 0.0f;
        }
        #pragma unroll
        for (int ct = 0; ct < 4; ct++)
            #pragma unroll
            for (int e = 0; e < 8; e++) {
                float p = __expf(sc[ct][e] - mnew[e]);
                sc[ct][e] = p;
                rs[e] += p;
            }
        #pragma unroll
        for (int mk = 1; mk < 16; mk <<= 1)
            #pragma unroll
            for (int e = 0; e < 8; e++) rs[e] += __shfl_xor(rs[e], mk, 32);
        #pragma unroll
        for (int e = 0; e < 8; e++) {
            l_run[e] = l_run[e] * alpha[e] + rs[e];
            m_run[e] = mnew[e];
        }
        #pragma unroll
        for (int nt = 0; nt < 4; nt++)
            #pragma unroll
            for (int e = 0; e < 8; e++) oacc[nt][e] *= alpha[e];

        // ---- write P (C-layout regs) into per-wave LDS as A-fragments ----
        int prow = 8 * (lane >> 4);
        #pragma unroll
        for (int ct = 0; ct < 4; ct++) {
            int col = ct * 16 + (lane & 15);
            int f   = col >> 5, kk = col & 31;
            int lp  = 16 * (kk >> 4);
            int ep  = kk & 15;
            #pragma unroll
            for (int e = 0; e < 8; e++)
                sP[w][f][(prow + e + lp) * 16 + ep] = (bf16)sc[ct][e];
        }
        v16bf pa0 = *(const v16bf*)&sP[w][0][lane * 16];
        v16bf pa1 = *(const v16bf*)&sP[w][1][lane * 16];
        #pragma unroll
        for (int nt = 0; nt < 4; nt++) {
            oacc[nt] = WMMA_BF16(pa0, *(const v16bf*)&sV[nt * 2 + 0][lane * 16], oacc[nt]);
            oacc[nt] = WMMA_BF16(pa1, *(const v16bf*)&sV[nt * 2 + 1][lane * 16], oacc[nt]);
        }
        __syncthreads();
    }

    // ---- normalize and write output ----
    int lcol = lane & 15, prow = 8 * (lane >> 4);
    #pragma unroll
    for (int nt = 0; nt < 4; nt++)
        #pragma unroll
        for (int e = 0; e < 8; e++) {
            size_t tokr = tok0 + w * 16 + prow + e;
            obuf[tokr * 768 + hh * 64 + nt * 16 + lcol] = (bf16)(oacc[nt][e] / l_run[e]);
        }
}

// ===========================================================================
// Elementwise / small kernels
// ===========================================================================
__global__ __launch_bounds__(256) void wt_tr(const float* __restrict__ in,
                                             bf16* __restrict__ out, int K, int N) {
    int i = blockIdx.x * 256 + threadIdx.x;
    if (i >= K * N) return;
    int k = i / N, nn = i % N;
    out[(size_t)nn * K + k] = (bf16)in[i];
}

__global__ __launch_bounds__(256) void emb_ln(
    const int* __restrict__ ids, const float* __restrict__ emb,
    const float* __restrict__ pos, const float* __restrict__ g,
    const float* __restrict__ bta, float* __restrict__ h, bf16* __restrict__ hb)
{
    __shared__ float red[256];
    int row = blockIdx.x;
    int s   = row & 4095;
    int id  = ids[row];
    float x[3]; float sum = 0.0f;
    #pragma unroll
    for (int i = 0; i < 3; i++) {
        int d = threadIdx.x + i * 256;
        x[i] = emb[(size_t)id * 768 + d] + pos[(size_t)s * 768 + d];
        sum += x[i];
    }
    red[threadIdx.x] = sum; __syncthreads();
    for (int o = 128; o > 0; o >>= 1) { if (threadIdx.x < o) red[threadIdx.x] += red[threadIdx.x + o]; __syncthreads(); }
    float mean = red[0] * (1.0f / 768.0f); __syncthreads();
    float vs = 0.0f;
    #pragma unroll
    for (int i = 0; i < 3; i++) { float dd = x[i] - mean; vs += dd * dd; }
    red[threadIdx.x] = vs; __syncthreads();
    for (int o = 128; o > 0; o >>= 1) { if (threadIdx.x < o) red[threadIdx.x] += red[threadIdx.x + o]; __syncthreads(); }
    float inv = rsqrtf(red[0] * (1.0f / 768.0f) + 1e-12f);
    #pragma unroll
    for (int i = 0; i < 3; i++) {
        int d = threadIdx.x + i * 256;
        float y = (x[i] - mean) * inv * g[d] + bta[d];
        h [(size_t)row * 768 + d] = y;
        hb[(size_t)row * 768 + d] = (bf16)y;
    }
}

__global__ __launch_bounds__(256) void res_ln(
    float* h, const float* __restrict__ add, const float* __restrict__ g,
    const float* __restrict__ bta, bf16* __restrict__ hb)
{
    __shared__ float red[256];
    int row = blockIdx.x;
    float x[3]; float sum = 0.0f;
    #pragma unroll
    for (int i = 0; i < 3; i++) {
        int d = threadIdx.x + i * 256;
        x[i] = h[(size_t)row * 768 + d] + add[(size_t)row * 768 + d];
        sum += x[i];
    }
    red[threadIdx.x] = sum; __syncthreads();
    for (int o = 128; o > 0; o >>= 1) { if (threadIdx.x < o) red[threadIdx.x] += red[threadIdx.x + o]; __syncthreads(); }
    float mean = red[0] * (1.0f / 768.0f); __syncthreads();
    float vs = 0.0f;
    #pragma unroll
    for (int i = 0; i < 3; i++) { float dd = x[i] - mean; vs += dd * dd; }
    red[threadIdx.x] = vs; __syncthreads();
    for (int o = 128; o > 0; o >>= 1) { if (threadIdx.x < o) red[threadIdx.x] += red[threadIdx.x + o]; __syncthreads(); }
    float inv = rsqrtf(red[0] * (1.0f / 768.0f) + 1e-12f);
    #pragma unroll
    for (int i = 0; i < 3; i++) {
        int d = threadIdx.x + i * 256;
        float y = (x[i] - mean) * inv * g[d] + bta[d];
        h [(size_t)row * 768 + d] = y;
        hb[(size_t)row * 768 + d] = (bf16)y;
    }
}

__global__ __launch_bounds__(256) void pooler(
    const float* __restrict__ h, const float* __restrict__ Wp,
    const float* __restrict__ bp, float* __restrict__ pooled)
{
    int i = blockIdx.x * 256 + threadIdx.x;     // 0..3071
    int b = i / 768, jcol = i % 768;
    const float* hc = h + (size_t)b * 4096 * 768;  // [CLS] row
    float s = bp[jcol];
    for (int d = 0; d < 768; d++) s += hc[d] * Wp[d * 768 + jcol];
    pooled[i] = tanhf(s);
}

__global__ __launch_bounds__(32) void classifier(
    const float* __restrict__ pooled, const float* __restrict__ Wc,
    const float* __restrict__ bc, const int* __restrict__ label,
    float* __restrict__ out)
{
    __shared__ float logits[8];
    int t = threadIdx.x;
    if (t < 8) {
        int b = t >> 1, c = t & 1;
        float s = bc[c];
        for (int d = 0; d < 768; d++) s += pooled[b * 768 + d] * Wc[d * 2 + c];
        logits[t] = s;
    }
    __syncthreads();
    if (t == 0) {
        float loss = 0.0f;
        for (int b = 0; b < 4; b++) {
            float l0 = logits[b * 2], l1 = logits[b * 2 + 1];
            float m  = fmaxf(l0, l1);
            float e0 = __expf(l0 - m), e1 = __expf(l1 - m);
            float p0 = e0 / (e0 + e1), p1 = e1 / (e0 + e1);
            out[b * 2] = p0; out[b * 2 + 1] = p1;
            // faithful double softmax: log_softmax applied to probs
            float mm = fmaxf(p0, p1);
            float z  = __expf(p0 - mm) + __expf(p1 - mm);
            float ls0 = p0 - mm - logf(z), ls1 = p1 - mm - logf(z);
            loss += (label[b] == 0) ? -ls0 : -ls1;
        }
        out[8] = loss * 0.25f;
    }
}

// ===========================================================================
// Launch orchestration
// ===========================================================================
extern "C" void kernel_launch(void* const* d_in, const int* in_sizes, int n_in,
                              void* d_out, int out_size, void* d_ws, size_t ws_size,
                              hipStream_t stream) {
    (void)in_sizes; (void)n_in; (void)out_size; (void)ws_size;

    const int*   ids  = (const int*)  d_in[0];
    const int*   am   = (const int*)  d_in[1];
    const int*   lbl  = (const int*)  d_in[2];
    const int*   ridx = (const int*)  d_in[3];
    const float* emb  = (const float*)d_in[4];
    const float* pos  = (const float*)d_in[5];
    const float* lng  = (const float*)d_in[6];
    const float* lnb  = (const float*)d_in[7];
    const float* Wqkv = (const float*)d_in[8];
    const float* bqkv = (const float*)d_in[9];
    const float* Wo   = (const float*)d_in[10];
    const float* bo   = (const float*)d_in[11];
    const float* ln1g = (const float*)d_in[12];
    const float* ln1b = (const float*)d_in[13];
    const float* Wff1 = (const float*)d_in[14];
    const float* bff1 = (const float*)d_in[15];
    const float* Wff2 = (const float*)d_in[16];
    const float* bff2 = (const float*)d_in[17];
    const float* ln2g = (const float*)d_in[18];
    const float* ln2b = (const float*)d_in[19];
    const float* Wp   = (const float*)d_in[20];
    const float* bp   = (const float*)d_in[21];
    const float* Wc   = (const float*)d_in[22];
    const float* bc   = (const float*)d_in[23];

    const size_t M = 16384;   // B*S
    char* p = (char*)d_ws;
    float* h     = (float*)p;  p += M * 768 * 4;
    bf16*  hb    = (bf16*) p;  p += M * 768 * 2;
    float* tmp   = (float*)p;  p += M * 768 * 4;
    bf16*  obuf  = (bf16*) p;  p += M * 768 * 2;
    bf16*  qkvb  = (bf16*) p;  p += M * 2304 * 2;
    bf16*  ff1b  = (bf16*) p;  p += M * 3072 * 2;
    bf16*  WqkvT = (bf16*) p;  p += (size_t)2 * 2304 * 768 * 2;
    bf16*  WoT   = (bf16*) p;  p += (size_t)2 * 768 * 768 * 2;
    bf16*  Wff1T = (bf16*) p;  p += (size_t)2 * 3072 * 768 * 2;
    bf16*  Wff2T = (bf16*) p;  p += (size_t)2 * 768 * 3072 * 2;
    float* pooledb = (float*)p; p += 4 * 768 * 4;

    // Weight conversion: fp32 [K][N] -> bf16 [N][K]
    for (int l = 0; l < 2; l++) {
        int n1 = 768 * 2304;
        wt_tr<<<(n1 + 255) / 256, 256, 0, stream>>>(Wqkv + (size_t)l * n1, WqkvT + (size_t)l * n1, 768, 2304);
        int n2 = 768 * 768;
        wt_tr<<<(n2 + 255) / 256, 256, 0, stream>>>(Wo + (size_t)l * n2, WoT + (size_t)l * n2, 768, 768);
        int n3 = 768 * 3072;
        wt_tr<<<(n3 + 255) / 256, 256, 0, stream>>>(Wff1 + (size_t)l * n3, Wff1T + (size_t)l * n3, 768, 3072);
        wt_tr<<<(n3 + 255) / 256, 256, 0, stream>>>(Wff2 + (size_t)l * n3, Wff2T + (size_t)l * n3, 3072, 768);
    }

    emb_ln<<<16384, 256, 0, stream>>>(ids, emb, pos, lng, lnb, h, hb);

    for (int l = 0; l < 2; l++) {
        gemm_bf16_wmma<<<dim3(18, 128), 256, 0, stream>>>(
            hb, WqkvT + (size_t)l * 2304 * 768, bqkv + l * 2304, qkvb, nullptr,
            16384, 2304, 768, 0);
        bigbird_attn<<<4 * 12 * 64, 128, 0, stream>>>(qkvb, am, ridx, obuf);
        gemm_bf16_wmma<<<dim3(6, 128), 256, 0, stream>>>(
            obuf, WoT + (size_t)l * 768 * 768, bo + l * 768, nullptr, tmp,
            16384, 768, 768, 0);
        res_ln<<<16384, 256, 0, stream>>>(h, tmp, ln1g + l * 768, ln1b + l * 768, hb);
        gemm_bf16_wmma<<<dim3(24, 128), 256, 0, stream>>>(
            hb, Wff1T + (size_t)l * 3072 * 768, bff1 + l * 3072, ff1b, nullptr,
            16384, 3072, 768, 1);
        gemm_bf16_wmma<<<dim3(6, 128), 256, 0, stream>>>(
            ff1b, Wff2T + (size_t)l * 768 * 3072, bff2 + l * 768, nullptr, tmp,
            16384, 768, 3072, 0);
        res_ln<<<16384, 256, 0, stream>>>(h, tmp, ln2g + l * 768, ln2b + l * 768, hb);
    }

    pooler<<<12, 256, 0, stream>>>(h, Wp, bp, pooledb);
    classifier<<<1, 32, 0, stream>>>(pooledb, Wc, bc, lbl, (float*)d_out);
}
